// PPM_27917287424696
// MI455X (gfx1250) — matrix operand order repaired and checked by reading, not compiled, not run on hardware
//
#include <hip/hip_runtime.h>
#include <hip/hip_bf16.h>
#include <stdint.h>

typedef __attribute__((ext_vector_type(16))) __bf16 v16bf;
typedef __attribute__((ext_vector_type(8)))  float  v8f;
typedef __attribute__((ext_vector_type(8)))  __bf16 bf16x8;
typedef __attribute__((ext_vector_type(4)))  __bf16 bf16x4;
typedef __attribute__((ext_vector_type(2)))  int    i32x2;
typedef __attribute__((ext_vector_type(4)))  int    i32x4;

namespace {
constexpr int C  = 256;   // channels
constexpr int HW = 4096;  // pixels per batch (64x64)
constexpr int NB = 8;     // batch
constexpr float EPS = 1e-12f;
}

// ---------------------------------------------------------------------------
// gfx1250 async global->LDS path (ASYNCcnt). Builtins take typed vector
// pointers: (vNi addrspace(1)* src, vNi addrspace(3)* dst, imm off, imm cpol).
// Guarded so the file still compiles if a toolchain lacks the builtins.
// ---------------------------------------------------------------------------
#if defined(__HIP_DEVICE_COMPILE__) && \
    __has_builtin(__builtin_amdgcn_global_load_async_to_lds_b64)
#define HAVE_ASYNC_LDS 1
#else
#define HAVE_ASYNC_LDS 0
#endif
#if defined(__HIP_DEVICE_COMPILE__) && \
    __has_builtin(__builtin_amdgcn_global_load_async_to_lds_b128)
#define HAVE_ASYNC_LDS_128 1
#else
#define HAVE_ASYNC_LDS_128 0
#endif

typedef __attribute__((address_space(1))) i32x2 g_i32x2;
typedef __attribute__((address_space(3))) i32x2 l_i32x2;
typedef __attribute__((address_space(1))) i32x4 g_i32x4;
typedef __attribute__((address_space(3))) i32x4 l_i32x4;

__device__ __forceinline__ void copy_b64_to_lds(const __bf16* g, __bf16* l) {
#if HAVE_ASYNC_LDS
  // one instruction: LDS[l] = MEM[g] (8 bytes per lane), tracked by ASYNCcnt
  __builtin_amdgcn_global_load_async_to_lds_b64(
      (g_i32x2*)(uintptr_t)g, (l_i32x2*)l, 0, 0);
#else
  *(bf16x4*)l = *(const bf16x4*)g;
#endif
}

__device__ __forceinline__ void copy_b128_to_lds(const __bf16* g, __bf16* l) {
#if HAVE_ASYNC_LDS_128
  // one instruction: LDS[l] = MEM[g] (16 bytes per lane), tracked by ASYNCcnt
  __builtin_amdgcn_global_load_async_to_lds_b128(
      (g_i32x4*)(uintptr_t)g, (l_i32x4*)l, 0, 0);
#elif HAVE_ASYNC_LDS
  __builtin_amdgcn_global_load_async_to_lds_b64(
      (g_i32x2*)(uintptr_t)g, (l_i32x2*)l, 0, 0);
  __builtin_amdgcn_global_load_async_to_lds_b64(
      (g_i32x2*)((uintptr_t)g + 8), (l_i32x2*)(l + 4), 0, 0);
#else
  *(bf16x8*)l = *(const bf16x8*)g;
#endif
}

__device__ __forceinline__ void async_copy_wait() {
#if HAVE_ASYNC_LDS || HAVE_ASYNC_LDS_128
#if __has_builtin(__builtin_amdgcn_s_wait_asynccnt)
  __builtin_amdgcn_s_wait_asynccnt(0);
#else
  asm volatile("s_wait_asynccnt 0x0" ::: "memory");
#endif
#endif
}

// ---------------------------------------------------------------------------
// Fragment-order LDS layout (wave32, ISA 7.12.2):
// every 16x16(x32) WMMA operand fragment occupies 512 bf16 (1 KB); each lane's
// 16 values are contiguous (32 B) so a fragment load is 2x ds_load_b128.
//   A-frag (16 rows x 32 K): lane = (m&15) + ((k>>3)&1)*16, elem = (k>>4)*8+(k&7)
//   B-frag (32 K x 16 cols): lane = (n&15) + ((k>>3)&1)*16, elem = (k>>4)*8+(k&7)
// C/D 16x16 f32: VGPR v holds row M = v + (lane>=16 ? 8 : 0), col N = lane&15.
// Note: any 8-aligned K-run of 8 elems maps to 16 contiguous bytes in one
// lane's region -> async b128 staging hits fragment order directly.
// ---------------------------------------------------------------------------
__device__ __forceinline__ v16bf load_frag(const __bf16* fragbase) {
  return *(const v16bf*)(fragbase + (threadIdx.x & 31) * 16);
}

// ---------------------------------------------------------------------------
// Kernel 0: W (fp32 256x256) -> bf16
// ---------------------------------------------------------------------------
__global__ void k_convert_w(const float* __restrict__ W, __bf16* __restrict__ w_bf) {
  const int i = blockIdx.x * blockDim.x + threadIdx.x;  // exactly C*C threads
  w_bf[i] = (__bf16)W[i];
}

// ---------------------------------------------------------------------------
// Kernel 1: per-pixel L2 norm over channels; xn = x / max(norm, eps) in bf16,
// normv = max(norm, eps) (so xn * normv == x for the transform step).
// ---------------------------------------------------------------------------
__global__ void k_normalize(const float* __restrict__ x, __bf16* __restrict__ xn,
                            float* __restrict__ normv) {
  const int p = blockIdx.x * blockDim.x + threadIdx.x;  // exactly NB*HW threads
  const int b = p >> 12;
  const int i = p & (HW - 1);
  const float* xp = x + (size_t)b * C * HW + i;
  float ss = 0.f;
#pragma unroll 8
  for (int c = 0; c < C; ++c) {
    const float v = xp[(size_t)c * HW];
    ss += v * v;
  }
  const float m = fmaxf(sqrtf(ss), EPS);
  const float inv = 1.0f / m;
  normv[p] = m;
  __bf16* xo = xn + (size_t)b * C * HW + i;
#pragma unroll 8
  for (int c = 0; c < C; ++c)
    xo[(size_t)c * HW] = (__bf16)(xp[(size_t)c * HW] * inv);
}

// ---------------------------------------------------------------------------
// Kernel 2: t = (W @ xn) * diag(normv) + bias, output bf16 [b][c][hw].
// Block: (batch, 64-pixel tile), 8 waves, each wave owns a 32x64 output strip.
// ---------------------------------------------------------------------------
__global__ __launch_bounds__(256) void k_transform(
    const __bf16* __restrict__ xn, const __bf16* __restrict__ w_bf,
    const float* __restrict__ bias, const float* __restrict__ normv,
    __bf16* __restrict__ t_bf) {
  __shared__ __align__(16) __bf16 lds_w[C * 32];   // 16 A-frags (c-major), 16 KB
  __shared__ __align__(16) __bf16 lds_x[32 * 64];  // 4 B-frags (n-major),  4 KB

  const int b    = blockIdx.x >> 6;
  const int i0   = (blockIdx.x & 63) * 64;
  const int tid  = threadIdx.x;
  const int wave = tid >> 5;
  const int lane = tid & 31;

  v8f acc[2][4] = {};

  for (int k0 = 0; k0 < C; k0 += 32) {
    // stage W tile -> A-fragment order (async global->LDS, 16 bytes per op)
#pragma unroll
    for (int g = 0; g < 4; ++g) {
      const int e  = (tid + g * 256) * 8;  // flat elem in 256x32 tile
      const int c  = e >> 5;
      const int k8 = e & 31;               // 0, 8, 16, 24
      const int frag = (c >> 4) * 512;
      const int ln   = (c & 15) + ((k8 >> 3) & 1) * 16;
      const int ee   = (k8 >> 4) << 3;
      copy_b128_to_lds(&w_bf[(size_t)c * C + k0 + k8], &lds_w[frag + ln * 16 + ee]);
    }
    // stage xn tile -> B-fragment order (transpose: scatter on LDS-store side)
#pragma unroll
    for (int g = 0; g < 2; ++g) {
      const int e  = (tid + g * 256) * 4;  // flat elem in 32x64 tile
      const int kk = e >> 6;
      const int n4 = e & 63;
      const bf16x4 v = *(const bf16x4*)&xn[(size_t)(b * C + k0 + kk) * HW + i0 + n4];
      const int frag = (n4 >> 4) * 512;
      const int lk   = ((kk >> 3) & 1) * 16;
      const int ee   = ((kk >> 4) << 3) + (kk & 7);
#pragma unroll
      for (int q = 0; q < 4; ++q)
        lds_x[frag + (((n4 + q) & 15) + lk) * 16 + ee] = v[q];
    }
    async_copy_wait();
    __syncthreads();
#pragma unroll
    for (int sc = 0; sc < 2; ++sc) {
      const v16bf a = load_frag(lds_w + (wave * 2 + sc) * 512);
#pragma unroll
      for (int sj = 0; sj < 4; ++sj) {
        const v16bf bb = load_frag(lds_x + sj * 512);
        acc[sc][sj] = __builtin_amdgcn_wmma_f32_16x16x32_bf16(
            false, a, false, bb, (short)0, acc[sc][sj], false, false);
      }
    }
    __syncthreads();
  }

  // epilogue: scale by per-pixel norm, add bias, cast bf16
#pragma unroll
  for (int sc = 0; sc < 2; ++sc)
#pragma unroll
    for (int sj = 0; sj < 4; ++sj)
#pragma unroll
      for (int v = 0; v < 8; ++v) {
        const int c = wave * 32 + sc * 16 + v + ((lane >= 16) ? 8 : 0);
        const int j = i0 + sj * 16 + (lane & 15);
        const float val = acc[sc][sj][v] * normv[b * HW + j] + bias[c];
        t_bf[(size_t)(b * C + c) * HW + j] = (__bf16)val;
      }
}

// ---------------------------------------------------------------------------
// Kernel 3 (main): out[:, j] = sum_i t[:, i] * (xn_i . xn_j)^2, fused.
// Block: (batch, 64-col j-tile). Xn_j stays resident in LDS (B-frag order);
// per 32-row i-tile: stage A computes the 32x64 S tile via WMMA, squares and
// packs it straight into B-fragment order (one ds_store_b128 per lane);
// stage B accumulates the 256x64 output. lds_a is time-shared between
// Xn_i (phase A, VGPR transpose) and T_i (phase B, async b128 global->LDS).
// ---------------------------------------------------------------------------
__global__ __launch_bounds__(256) void k_propagate(
    const __bf16* __restrict__ xn, const __bf16* __restrict__ t_bf,
    float* __restrict__ out) {
  __shared__ __align__(16) __bf16 lds_xnj[C * 64];  // 32 B-frags, 32 KB (resident)
  __shared__ __align__(16) __bf16 lds_a[C * 32];    // 16 A-frags, 16 KB (time-shared)
  __shared__ __align__(16) __bf16 lds_s[32 * 64];   // 4 B-frags, 4 KB

  const int b    = blockIdx.x >> 6;
  const int j0   = (blockIdx.x & 63) * 64;
  const int tid  = threadIdx.x;
  const int wave = tid >> 5;
  const int lane = tid & 31;

  // resident Xn_j tile -> B-fragment order; frag grid (kb = c>>5, nj = n>>4)
#pragma unroll
  for (int g = 0; g < 16; ++g) {
    const int e  = (tid + g * 256) * 4;  // flat elem in 256x64 tile
    const int c  = e >> 6;
    const int n4 = e & 63;
    const bf16x4 v = *(const bf16x4*)&xn[(size_t)(b * C + c) * HW + j0 + n4];
    const int frag = ((c >> 5) * 4 + (n4 >> 4)) * 512;
    const int lk   = ((c >> 3) & 1) * 16;
    const int ee   = (((c & 31) >> 4) << 3) + (c & 7);
#pragma unroll
    for (int q = 0; q < 4; ++q)
      lds_xnj[frag + (((n4 + q) & 15) + lk) * 16 + ee] = v[q];
  }

  v8f acc[2][4] = {};

  for (int i0 = 0; i0 < HW; i0 += 32) {
    __syncthreads();  // lds_a free (prev stage B done); also covers lds_xnj init

    // phase A: lds_a = Xn_i as A-frags; frag grid (mi = m>>4, kb = c>>5)
    {
      const int m4 = (tid & 7) * 4;
      const int cb = (tid >> 3) * 8;
#pragma unroll
      for (int cc = 0; cc < 8; ++cc) {
        const int c = cb + cc;
        const bf16x4 v = *(const bf16x4*)&xn[(size_t)(b * C + c) * HW + i0 + m4];
        const int kl   = c & 31;
        const int frag = ((m4 >> 4) * 8 + (c >> 5)) * 512;
        const int lb   = ((kl >> 3) & 1) * 16;
        const int ee   = ((kl >> 4) << 3) + (kl & 7);
#pragma unroll
        for (int q = 0; q < 4; ++q)
          lds_a[frag + (((m4 + q) & 15) + lb) * 16 + ee] = v[q];
      }
    }
    // prefetch next i-tile (global_prefetch_b8)
    if (i0 + 32 < HW) {
      __builtin_prefetch(&t_bf[(size_t)(b * C + (tid >> 3)) * HW + i0 + 32], 0, 1);
      __builtin_prefetch(&xn[(size_t)(b * C + (tid >> 3)) * HW + i0 + 32], 0, 1);
    }
    __syncthreads();

    // stage A: S = (Xn_i^T Xn_j); one 16x16 tile per wave (2x4 grid), K = 256
    {
      const int si = wave & 1;
      const int sj = wave >> 1;
      v8f s = {};
#pragma unroll
      for (int kk = 0; kk < 8; ++kk) {
        const v16bf a  = load_frag(lds_a + (si * 8 + kk) * 512);
        const v16bf bb = load_frag(lds_xnj + (kk * 4 + sj) * 512);
        s = __builtin_amdgcn_wmma_f32_16x16x32_bf16(false, a, false, bb,
                                                    (short)0, s, false, false);
      }
      // relu(sim^2) == sim^2; D-layout maps to 8 contiguous bf16 of the
      // B-fragment -> single packed ds_store_b128 per lane
      bf16x8 sq;
#pragma unroll
      for (int v = 0; v < 8; ++v) {
        const float sv = s[v];
        sq[v] = (__bf16)(sv * sv);
      }
      const int dst = sj * 512 + ((lane & 15) + ((lane >= 16) ? 16 : 0)) * 16 + si * 8;
      *(bf16x8*)&lds_s[dst] = sq;
    }
    __syncthreads();

    // phase B: lds_a = T_i as A-frags (async b128 global->LDS, frag-order dest)
#pragma unroll
    for (int g = 0; g < 4; ++g) {
      const int e  = (tid + g * 256) * 8;  // flat elem in 256x32 tile
      const int c  = e >> 5;
      const int k8 = e & 31;               // 0, 8, 16, 24
      const int frag = (c >> 4) * 512;
      const int ln   = (c & 15) + ((k8 >> 3) & 1) * 16;
      const int ee   = (k8 >> 4) << 3;
      copy_b128_to_lds(&t_bf[(size_t)(b * C + c) * HW + i0 + k8],
                       &lds_a[frag + ln * 16 + ee]);
    }
    async_copy_wait();
    __syncthreads();

    // stage B: out(256x64) += T_i(256x32) @ S(32x64); wave owns 32 c-rows
#pragma unroll
    for (int sc = 0; sc < 2; ++sc) {
      const v16bf a = load_frag(lds_a + (wave * 2 + sc) * 512);
#pragma unroll
      for (int sj = 0; sj < 4; ++sj) {
        const v16bf bb = load_frag(lds_s + sj * 512);
        acc[sc][sj] = __builtin_amdgcn_wmma_f32_16x16x32_bf16(
            false, a, false, bb, (short)0, acc[sc][sj], false, false);
      }
    }
  }

  // write fp32 output
#pragma unroll
  for (int sc = 0; sc < 2; ++sc)
#pragma unroll
    for (int sj = 0; sj < 4; ++sj)
#pragma unroll
      for (int v = 0; v < 8; ++v) {
        const int c = wave * 32 + sc * 16 + v + ((lane >= 16) ? 8 : 0);
        const int j = j0 + sj * 16 + (lane & 15);
        out[(size_t)(b * C + c) * HW + j] = acc[sc][sj][v];
      }
}

// ---------------------------------------------------------------------------
extern "C" void kernel_launch(void* const* d_in, const int* in_sizes, int n_in,
                              void* d_out, int out_size, void* d_ws, size_t ws_size,
                              hipStream_t stream) {
  (void)in_sizes; (void)n_in; (void)out_size; (void)ws_size;
  const float* x    = (const float*)d_in[0];
  const float* W    = (const float*)d_in[1];
  const float* bias = (const float*)d_in[2];
  float* out = (float*)d_out;

  // workspace layout (~33.8 MB):
  //   xn_bf : NB*C*HW bf16 (16 MB)
  //   t_bf  : NB*C*HW bf16 (16 MB)
  //   w_bf  : C*C bf16     (128 KB)
  //   normv : NB*HW fp32   (128 KB)
  char* ws = (char*)d_ws;
  const size_t sz_feat = (size_t)NB * C * HW * sizeof(__bf16);
  __bf16* xn_bf = (__bf16*)ws;
  __bf16* t_bf  = (__bf16*)(ws + sz_feat);
  __bf16* w_bf  = (__bf16*)(ws + 2 * sz_feat);
  float*  normv = (float*)(ws + 2 * sz_feat + (size_t)C * C * sizeof(__bf16));

  k_convert_w<<<(C * C) / 256, 256, 0, stream>>>(W, w_bf);
  k_normalize<<<(NB * HW) / 256, 256, 0, stream>>>(x, xn_bf, normv);
  k_transform<<<NB * (HW / 64), 256, 0, stream>>>(xn_bf, w_bf, bias, normv, t_bf);
  k_propagate<<<NB * (HW / 64), 256, 0, stream>>>(xn_bf, t_bf, out);
}
